// GroupedQueryAttention_19533511262527
// MI455X (gfx1250) — compile-verified, matrix operand order checked by reading, not run on hardware
//
#include <hip/hip_runtime.h>

// ---------------------------------------------------------------------------
// GQA forward for MI455X (gfx1250, wave32, WMMA bf16 16x16x32,
// async global->LDS staging with double buffering).
// ---------------------------------------------------------------------------

typedef __attribute__((ext_vector_type(16))) __bf16 v16bf;
typedef __attribute__((ext_vector_type(8)))  float  v8f;

union FragBF { v16bf v; uint4 u[2]; };

#define B_   2
#define S_   2048
#define DIM_ 2048
#define H_   16
#define KV_  4
#define HD_  128
#define M_   (B_ * S_)   // 4096 rows (b*S+s)

__device__ __forceinline__ unsigned short f2bf(float f) {
  unsigned u = __float_as_uint(f);
  u += 0x7FFFu + ((u >> 16) & 1u);          // round-to-nearest-even
  return (unsigned short)(u >> 16);
}
__device__ __forceinline__ float bf2f(unsigned short h) {
  return __uint_as_float(((unsigned)h) << 16);
}

// CDNA5 async global->LDS copy (ASYNCcnt tracked), 16B per lane.
// GVS addressing: uniform 64-bit SGPR base + per-lane signed 32-bit byte off.
__device__ __forceinline__ void async_ld_b128(unsigned lds_off, const void* sbase, int voff) {
  asm volatile("global_load_async_to_lds_b128 %0, %1, %2"
               : : "v"(lds_off), "v"(voff), "s"(sbase) : "memory");
}
__device__ __forceinline__ void wait_async0() {
  asm volatile("s_wait_asynccnt 0x0" : : : "memory");
}
__device__ __forceinline__ unsigned lds_off_u32(const void* p) {
  return (unsigned)(uintptr_t)p;    // LDS aperture: addr[31:0] is the LDS offset
}

// ---------------------------------------------------------------------------
// Generic bf16 WMMA GEMM:  C[M,N] = A[M,K] * Bt[N,K]^T
// block = 256 thr (8 waves), tile 128x128, k-step 32, double-buffered async
// staging (one barrier per k-step). LDS rows padded to 40 bf16 (20 dwords)
// -> 16B frag loads start 4 dwords apart mod 64 banks : conflict-free.
// ---------------------------------------------------------------------------
template<bool F32OUT>
__global__ __launch_bounds__(256)
void gemm_bf16_wmma(const unsigned short* __restrict__ A,
                    const unsigned short* __restrict__ Bt,
                    void* __restrict__ Cout,
                    int M, int N, int K)
{
  __shared__ unsigned short sA[2][128 * 40];
  __shared__ unsigned short sB[2][128 * 40];

  const int t    = threadIdx.x;
  const int lane = t & 31;
  const int wave = t >> 5;
  const int wy   = wave >> 1;          // 4 wave-rows of 32
  const int wx   = wave & 1;           // 2 wave-cols of 64
  const int fr   = lane & 15;
  const int fh   = lane >> 4;
  const long bm  = (long)blockIdx.y * 128;
  const long bn  = (long)blockIdx.x * 128;

  const int frow = t >> 1;             // 128 rows, 2 threads/row
  const int fk   = (t & 1) * 16;       // 16 bf16 (32B) per thread

  v8f acc[2][4] = {};

  // per-thread byte offsets (fit in int: <= 16.8 MB)
  const int aoff = (int)(((bm + frow) * (long)K + fk) * 2);
  const int boff = (int)(((bn + frow) * (long)K + fk) * 2);
  const unsigned sAb = lds_off_u32(&sA[0][0]);
  const unsigned sBb = lds_off_u32(&sB[0][0]);
  const unsigned ldsA = sAb + (unsigned)(frow * 40 + fk) * 2;
  const unsigned ldsB = sBb + (unsigned)(frow * 40 + fk) * 2;

  auto issue = [&](int k0, int buf) {
    unsigned la = ldsA + (unsigned)buf * (128 * 40 * 2);
    unsigned lb = ldsB + (unsigned)buf * (128 * 40 * 2);
    async_ld_b128(la,      A,  aoff + k0 * 2);
    async_ld_b128(la + 16, A,  aoff + k0 * 2 + 16);
    async_ld_b128(lb,      Bt, boff + k0 * 2);
    async_ld_b128(lb + 16, Bt, boff + k0 * 2 + 16);
  };

  issue(0, 0);
  int cur = 0;
  for (int k0 = 0; k0 < K; k0 += 32) {
    wait_async0();          // my async loads for buf[cur] are in LDS
    __syncthreads();        // everyone's loads done, prev compute done
    if (k0 + 32 < K) issue(k0 + 32, cur ^ 1);

    const unsigned short* cA = sA[cur];
    const unsigned short* cB = sB[cur];
    FragBF fa[2], fb[4];
#pragma unroll
    for (int mf = 0; mf < 2; ++mf) {
      int row = wy * 32 + mf * 16 + fr;
      fa[mf].u[0] = *(const uint4*)(cA + row * 40 + fh * 8);
      fa[mf].u[1] = *(const uint4*)(cA + row * 40 + fh * 8 + 16);
    }
#pragma unroll
    for (int nf = 0; nf < 4; ++nf) {
      int row = wx * 64 + nf * 16 + fr;
      fb[nf].u[0] = *(const uint4*)(cB + row * 40 + fh * 8);
      fb[nf].u[1] = *(const uint4*)(cB + row * 40 + fh * 8 + 16);
    }
#pragma unroll
    for (int mf = 0; mf < 2; ++mf)
#pragma unroll
      for (int nf = 0; nf < 4; ++nf)
        acc[mf][nf] = __builtin_amdgcn_wmma_f32_16x16x32_bf16(
            false, fa[mf].v, false, fb[nf].v, (short)0, acc[mf][nf], false, false);
    cur ^= 1;
  }

#pragma unroll
  for (int mf = 0; mf < 2; ++mf)
#pragma unroll
    for (int nf = 0; nf < 4; ++nf)
#pragma unroll
      for (int r = 0; r < 8; ++r) {
        long row = bm + wy * 32 + mf * 16 + r + 8 * fh;
        long col = bn + wx * 64 + nf * 16 + fr;
        if (F32OUT) ((float*)Cout)[row * N + col] = acc[mf][nf][r];
        else ((unsigned short*)Cout)[row * N + col] = f2bf(acc[mf][nf][r]);
      }
}

// ---------------------------------------------------------------------------
// Flash attention, 4 waves / block, 64 q-rows per block, 64-key tiles,
// double-buffered async K/V staging, base-2 online softmax.
// Q : bf16 [B][H][S][HD]  (A-frags, registers; 1/sqrt(HD)*log2(e) prefolded)
// K : bf16 [B][KV][S][HD] (row-major == B-frag [N=key][K=hd])
// Vt: bf16 [B][KV][HD][S] (row-major == B-frag [N=hd][K=key])
// ctx: bf16 [B*S][H*HD]
// ---------------------------------------------------------------------------
#define KSTR 136   // 68-dword stride -> conflict-free 16B frag loads
#define VSTR 72    // 36-dword stride -> conflict-free
#define PSTR 72

__global__ __launch_bounds__(128)
void attn_fwd_wmma(const unsigned short* __restrict__ Q,
                   const unsigned short* __restrict__ Kc,
                   const unsigned short* __restrict__ Vt,
                   unsigned short* __restrict__ ctx)
{
  __shared__ unsigned short sK[2][64 * KSTR];
  __shared__ unsigned short sV[2][128 * VSTR];
  __shared__ unsigned short sP[4 * 16 * PSTR];

  const int t    = threadIdx.x;
  const int lane = t & 31;
  const int wave = t >> 5;
  const int fr   = lane & 15;
  const int fh   = lane >> 4;

  const int bid  = blockIdx.x;
  const int qblk = bid & 31;            // S/64 = 32
  const int h    = (bid >> 5) & 15;
  const int b    = bid >> 9;
  const int kv   = h >> 2;              // REP = 4

  const int q0   = qblk * 64;
  const int qrow = q0 + wave * 16 + fr;

  // Q fragments (16 rows x HD=128 -> 4 A-frags of k=32), held in registers.
  const unsigned short* qptr = Q + (((long)(b * H_ + h)) * S_ + qrow) * HD_;
  FragBF aq[4];
#pragma unroll
  for (int j = 0; j < 4; ++j) {
    aq[j].u[0] = *(const uint4*)(qptr + j * 32 + fh * 8);
    aq[j].u[1] = *(const uint4*)(qptr + j * 32 + fh * 8 + 16);
  }

  float mrow[8], lrow[8];
#pragma unroll
  for (int r = 0; r < 8; ++r) { mrow[r] = -1e30f; lrow[r] = 0.f; }
  v8f o[8] = {};

  const unsigned short* kbase = Kc + ((long)(b * KV_ + kv)) * S_ * HD_;
  const unsigned short* vbase = Vt + ((long)(b * KV_ + kv)) * HD_ * S_;
  unsigned short* pmine = sP + wave * 16 * PSTR;

  const unsigned sKb = lds_off_u32(&sK[0][0]);
  const unsigned sVb = lds_off_u32(&sV[0][0]);

  auto issueTile = [&](int k0, int buf) {
    unsigned kb = sKb + (unsigned)buf * (64 * KSTR * 2);
    unsigned vb = sVb + (unsigned)buf * (128 * VSTR * 2);
#pragma unroll
    for (int i = 0; i < 8; ++i) {              // K tile: 64 x 128
      int id = t + 128 * i;
      int row = id >> 4, c = id & 15;
      async_ld_b128(kb + (unsigned)(row * KSTR + c * 8) * 2, kbase,
                    ((k0 + row) * HD_ + c * 8) * 2);
    }
#pragma unroll
    for (int i = 0; i < 8; ++i) {              // Vt tile: 128 x 64
      int id = t + 128 * i;
      int row = id >> 3, c = id & 7;
      async_ld_b128(vb + (unsigned)(row * VSTR + c * 8) * 2, vbase,
                    (row * S_ + k0 + c * 8) * 2);
    }
  };

  issueTile(0, 0);
  int cur = 0;
  for (int kt = 0; kt <= qblk; ++kt) {
    const int k0 = kt * 64;
    wait_async0();
    __syncthreads();
    if (kt < qblk) issueTile(k0 + 64, cur ^ 1);

    const unsigned short* cK = sK[cur];
    const unsigned short* cV = sV[cur];

    // S = (Q K^T) * (log2e/sqrt(HD))  (scale prefolded into Q)
    v8f sc[4] = {};
#pragma unroll
    for (int nf = 0; nf < 4; ++nf) {
      int krow = nf * 16 + fr;
#pragma unroll
      for (int j = 0; j < 4; ++j) {
        FragBF bk;
        bk.u[0] = *(const uint4*)(cK + krow * KSTR + j * 32 + fh * 8);
        bk.u[1] = *(const uint4*)(cK + krow * KSTR + j * 32 + fh * 8 + 16);
        sc[nf] = __builtin_amdgcn_wmma_f32_16x16x32_bf16(
            false, aq[j].v, false, bk.v, (short)0, sc[nf], false, false);
      }
    }

    if (kt == qblk) {                       // causal mask on diagonal tile
#pragma unroll
      for (int nf = 0; nf < 4; ++nf) {
        int key = k0 + nf * 16 + fr;
#pragma unroll
        for (int r = 0; r < 8; ++r) {
          int qq = q0 + wave * 16 + r + 8 * fh;
          if (key > qq) sc[nf][r] = -1e30f;
        }
      }
    }

    // base-2 online softmax; row r+8*fh lives in VGPR r across 16 lanes.
#pragma unroll
    for (int r = 0; r < 8; ++r) {
      float vmax = sc[0][r];
#pragma unroll
      for (int nf = 1; nf < 4; ++nf) vmax = fmaxf(vmax, sc[nf][r]);
      vmax = fmaxf(vmax, __shfl_xor(vmax, 8));
      vmax = fmaxf(vmax, __shfl_xor(vmax, 4));
      vmax = fmaxf(vmax, __shfl_xor(vmax, 2));
      vmax = fmaxf(vmax, __shfl_xor(vmax, 1));
      float mnew  = fmaxf(mrow[r], vmax);
      float alpha = exp2f(mrow[r] - mnew);        // v_exp_f32 directly
      float psum = 0.f;
#pragma unroll
      for (int nf = 0; nf < 4; ++nf) {
        float p = exp2f(sc[nf][r] - mnew);
        psum += p;
        pmine[(r + 8 * fh) * PSTR + nf * 16 + fr] = f2bf(p);  // C->A relayout
      }
      psum += __shfl_xor(psum, 8);
      psum += __shfl_xor(psum, 4);
      psum += __shfl_xor(psum, 2);
      psum += __shfl_xor(psum, 1);
      lrow[r] = lrow[r] * alpha + psum;
      mrow[r] = mnew;
#pragma unroll
      for (int f = 0; f < 8; ++f) o[f][r] *= alpha;
    }

    // P as two A-frags (16x64, k=32 each)
    FragBF pa[2];
    {
      const unsigned short* pp = pmine + fr * PSTR + fh * 8;
      pa[0].u[0] = *(const uint4*)(pp);
      pa[0].u[1] = *(const uint4*)(pp + 16);
      pa[1].u[0] = *(const uint4*)(pp + 32);
      pa[1].u[1] = *(const uint4*)(pp + 48);
    }
    // O += P @ V  (B-frag from Vt [hd][key])
#pragma unroll
    for (int f = 0; f < 8; ++f) {
      int vrow = f * 16 + fr;
      FragBF bv0, bv1;
      bv0.u[0] = *(const uint4*)(cV + vrow * VSTR + fh * 8);
      bv0.u[1] = *(const uint4*)(cV + vrow * VSTR + fh * 8 + 16);
      bv1.u[0] = *(const uint4*)(cV + vrow * VSTR + fh * 8 + 32);
      bv1.u[1] = *(const uint4*)(cV + vrow * VSTR + fh * 8 + 48);
      o[f] = __builtin_amdgcn_wmma_f32_16x16x32_bf16(
          false, pa[0].v, false, bv0.v, (short)0, o[f], false, false);
      o[f] = __builtin_amdgcn_wmma_f32_16x16x32_bf16(
          false, pa[1].v, false, bv1.v, (short)0, o[f], false, false);
    }
    cur ^= 1;
  }

  float inv[8];
#pragma unroll
  for (int r = 0; r < 8; ++r) inv[r] = 1.0f / lrow[r];
#pragma unroll
  for (int f = 0; f < 8; ++f)
#pragma unroll
    for (int r = 0; r < 8; ++r) {
      long row = (long)b * S_ + (q0 + wave * 16 + r + 8 * fh);
      ctx[row * (H_ * HD_) + h * HD_ + f * 16 + fr] = f2bf(o[f][r] * inv[r]);
    }
}

// ---------------------------------------------------------------------------
// Epilogue: per-head rmsnorm + RoPE; one wave per (b,s,head).
// in : bf16 [b*S+s][HH*HD] (gemm output); out : bf16 [B][HH][S][HD]
// ---------------------------------------------------------------------------
template<int HH>
__global__ __launch_bounds__(256)
void norm_rope_bf16(const unsigned short* __restrict__ in,
                    const float* __restrict__ g,
                    const float* __restrict__ cosb,
                    const float* __restrict__ sinb,
                    unsigned short* __restrict__ out, float scale)
{
  const int lane = threadIdx.x & 31;
  const int wave = threadIdx.x >> 5;
  long idx = (long)blockIdx.x * 8 + wave;   // (b*S+s)*HH + h
  int  h  = (int)(idx % HH);
  long bs = idx / HH;
  int  s  = (int)(bs % S_);
  int  b  = (int)(bs / S_);

  const unsigned short* row = in + bs * (long)(HH * HD_) + h * HD_;
  float x[4]; float ss = 0.f;
#pragma unroll
  for (int j = 0; j < 4; ++j) { x[j] = bf2f(row[lane + 32 * j]); ss += x[j] * x[j]; }
  ss += __shfl_xor(ss, 16); ss += __shfl_xor(ss, 8); ss += __shfl_xor(ss, 4);
  ss += __shfl_xor(ss, 2);  ss += __shfl_xor(ss, 1);
  float rinv = rsqrtf(ss * (1.0f / HD_) + 1e-6f);

  float xn[4];
#pragma unroll
  for (int j = 0; j < 4; ++j) { int d = lane + 32 * j; xn[j] = x[j] * rinv * g[d]; }

  unsigned short* orow = out + (((long)(b * HH + h)) * S_ + s) * HD_;
#pragma unroll
  for (int j = 0; j < 4; ++j) {
    int d = lane + 32 * j;                         // d and d+-64 are j<->j+2
    float rot = (j < 2) ? -xn[j + 2] : xn[j - 2];
    float val = (xn[j] * cosb[(long)s * HD_ + d] + rot * sinb[(long)s * HD_ + d]) * scale;
    orow[d] = f2bf(val);
  }
}

// V head split + transpose: bf16 [b*S+s][KV*HD] -> bf16 [B][KV][HD][S]
__global__ void v_head_transpose(const unsigned short* __restrict__ in,
                                 unsigned short* __restrict__ out)
{
  long i = (long)blockIdx.x * blockDim.x + threadIdx.x;
  long total = (long)B_ * S_ * KV_ * HD_;
  if (i >= total) return;
  int  d  = (int)(i % HD_);
  long r  = i / HD_;
  int  kv = (int)(r % KV_);
  long bs = r / KV_;
  int  s  = (int)(bs % S_);
  int  b  = (int)(bs / S_);
  out[(((long)(b * KV_ + kv)) * HD_ + d) * S_ + s] = in[i];
}

// f32 [K][N] -> bf16 [N][K] (so GEMM B-fragments load row-major [N][K])
__global__ __launch_bounds__(256)
void transpose_cvt_f32_bf16(const float* __restrict__ in,
                            unsigned short* __restrict__ out, int K, int N)
{
  __shared__ float tile[32][33];
  int nb = blockIdx.x * 32, kb = blockIdx.y * 32;
  int tx = threadIdx.x & 31, ty = threadIdx.x >> 5;
#pragma unroll
  for (int j = 0; j < 32; j += 8)
    tile[ty + j][tx] = in[(long)(kb + ty + j) * N + nb + tx];
  __syncthreads();
#pragma unroll
  for (int j = 0; j < 32; j += 8)
    out[(long)(nb + ty + j) * K + kb + tx] = f2bf(tile[tx][ty + j]);
}

__global__ void cvt_f32_bf16(const float* __restrict__ in,
                             unsigned short* __restrict__ out, long n)
{
  long i = (long)blockIdx.x * blockDim.x + threadIdx.x;
  if (i < n) out[i] = f2bf(in[i]);
}

// ---------------------------------------------------------------------------
extern "C" void kernel_launch(void* const* d_in, const int* in_sizes, int n_in,
                              void* d_out, int out_size, void* d_ws, size_t ws_size,
                              hipStream_t stream)
{
  (void)in_sizes; (void)n_in; (void)out_size; (void)ws_size;

  const float* x  = (const float*)d_in[0];
  const float* Wq = (const float*)d_in[1];
  const float* Wk = (const float*)d_in[2];
  const float* Wv = (const float*)d_in[3];
  const float* Wo = (const float*)d_in[4];
  const float* qg = (const float*)d_in[5];
  const float* kg = (const float*)d_in[6];
  const float* cs = (const float*)d_in[7];
  const float* sn = (const float*)d_in[8];
  // d_in[9] = mask : causal handled analytically
  float* out = (float*)d_out;

  char* w = (char*)d_ws;
  size_t off = 0;
  auto alloc = [&](size_t bytes) -> void* {
    void* p = w + off; off += (bytes + 255) & ~(size_t)255; return p;
  };
  unsigned short* xb   = (unsigned short*)alloc((size_t)M_ * DIM_ * 2);
  unsigned short* wqT  = (unsigned short*)alloc((size_t)DIM_ * DIM_ * 2);
  unsigned short* wkT  = (unsigned short*)alloc((size_t)512 * DIM_ * 2);
  unsigned short* wvT  = (unsigned short*)alloc((size_t)512 * DIM_ * 2);
  unsigned short* woT  = (unsigned short*)alloc((size_t)DIM_ * DIM_ * 2);
  unsigned short* qraw = (unsigned short*)alloc((size_t)M_ * DIM_ * 2);
  unsigned short* kraw = (unsigned short*)alloc((size_t)M_ * 512 * 2);
  unsigned short* vraw = (unsigned short*)alloc((size_t)M_ * 512 * 2);
  unsigned short* qh   = (unsigned short*)alloc((size_t)M_ * DIM_ * 2);
  unsigned short* kh   = (unsigned short*)alloc((size_t)M_ * 512 * 2);
  unsigned short* vt   = (unsigned short*)alloc((size_t)M_ * 512 * 2);
  unsigned short* ctx  = qraw;   // qraw dead after norm_rope -> reuse

  // 1) precision conversion + weight transposes (one-time, bandwidth-trivial)
  cvt_f32_bf16<<<(unsigned)(((long)M_ * DIM_) / 256), 256, 0, stream>>>(x, xb, (long)M_ * DIM_);
  transpose_cvt_f32_bf16<<<dim3(DIM_ / 32, DIM_ / 32), 256, 0, stream>>>(Wq, wqT, DIM_, DIM_);
  transpose_cvt_f32_bf16<<<dim3(512 / 32,  DIM_ / 32), 256, 0, stream>>>(Wk, wkT, DIM_, 512);
  transpose_cvt_f32_bf16<<<dim3(512 / 32,  DIM_ / 32), 256, 0, stream>>>(Wv, wvT, DIM_, 512);
  transpose_cvt_f32_bf16<<<dim3(DIM_ / 32, DIM_ / 32), 256, 0, stream>>>(Wo, woT, DIM_, DIM_);

  // 2) QKV projections (WMMA, async-staged)
  gemm_bf16_wmma<false><<<dim3(DIM_ / 128, M_ / 128), 256, 0, stream>>>(xb, wqT, qraw, M_, DIM_, DIM_);
  gemm_bf16_wmma<false><<<dim3(512  / 128, M_ / 128), 256, 0, stream>>>(xb, wkT, kraw, M_, 512,  DIM_);
  gemm_bf16_wmma<false><<<dim3(512  / 128, M_ / 128), 256, 0, stream>>>(xb, wvT, vraw, M_, 512,  DIM_);

  // 3) rmsnorm + RoPE; q picks up 1/sqrt(HD) * log2(e) for base-2 softmax
  const float qscale = (float)(0.08838834764831845 * 1.4426950408889634);
  norm_rope_bf16<H_ ><<<(B_ * S_ * H_)  / 8, 256, 0, stream>>>(qraw, qg, cs, sn, qh, qscale);
  norm_rope_bf16<KV_><<<(B_ * S_ * KV_) / 8, 256, 0, stream>>>(kraw, kg, cs, sn, kh, 1.0f);
  v_head_transpose<<<(unsigned)(((long)M_ * 512) / 256), 256, 0, stream>>>(vraw, vt);

  // 4) causal flash attention (WMMA, async-staged) -> ctx [B*S][H*HD] bf16
  attn_fwd_wmma<<<B_ * H_ * (S_ / 64), 128, 0, stream>>>(qh, kh, vt, ctx);

  // 5) output projection (WMMA, f32 out)
  gemm_bf16_wmma<true><<<dim3(DIM_ / 128, M_ / 128), 256, 0, stream>>>(ctx, woT, out, M_, DIM_, DIM_);
}
